// GenePerceiverEncoder_88029649699162
// MI455X (gfx1250) — matrix operand order repaired and checked by reading, not compiled, not run on hardware
//
#include <hip/hip_runtime.h>

// ---------------------------------------------------------------------------
// CDNA5 (gfx1250) Perceiver encoder. All GEMMs + attention run on
// v_wmma_f32_16x16x32_bf16 (bf16 inputs, fp32 accumulate). wave32 only.
// GEMM uses 32x64 register blocking per wave: 8 WMMAs per 32-deep k-step.
// ---------------------------------------------------------------------------

typedef __attribute__((ext_vector_type(16))) __bf16 v16bf;
typedef __attribute__((ext_vector_type(8)))  float  v8f;

#define Bc  4
#define Nc  8192
#define Dc  512
#define Hc  8
#define Lc  256
#define BKS 4
#define DHc 64

__device__ __forceinline__ __bf16 f2bf(float f) {
    union { float f; unsigned u; } v; v.f = f;
    unsigned r = v.u + 0x7FFFu + ((v.u >> 16) & 1u);
    unsigned short h = (unsigned short)(r >> 16);
    __bf16 out;
    __builtin_memcpy(&out, &h, 2);
    return out;
}

__device__ __forceinline__ v8f wmma_bf16(v16bf a, v16bf b, v8f c) {
    return __builtin_amdgcn_wmma_f32_16x16x32_bf16(
        /*neg_a=*/false, a, /*neg_b=*/false, b,
        /*c_mod=*/(short)0, c, /*reuse_a=*/false, /*reuse_b=*/false);
}

// A fragment: 16x32 (MxK) bf16 from row-major matrix, leading dim lda.
// Layout: lanes 0-15 (M=lane) hold K=0..15; lanes 16-31 (M=lane-16) hold K=16..31.
__device__ __forceinline__ v16bf load_A_frag(const __bf16* A, int lda, int lane) {
    int m  = lane & 15;
    int kb = (lane < 16) ? 0 : 16;
    const __bf16* row = A + (size_t)m * lda;
    v16bf a;
#pragma unroll
    for (int j = 0; j < 4; ++j) {
        int k = kb + 2 * j;
        a[2 * j]     = row[k];
        a[2 * j + 1] = row[k + 1];
        a[8 + 2 * j]     = row[k + 8];
        a[8 + 2 * j + 1] = row[k + 9];
    }
    return a;
}

// B fragment: 32x16 (KxN) where B[k][n] = W[n][k] (W row-major [N,K], ld ldw).
// Contiguous k order, same as A: lanes 0-15 (N=lane) hold K=0..15; lanes 16-31 hold K=16..31.
__device__ __forceinline__ v16bf load_Bt_frag(const __bf16* W, int ldw, int lane) {
    int n  = lane & 15;
    int kb = (lane < 16) ? 0 : 16;
    const __bf16* row = W + (size_t)n * ldw;
    v16bf b;
#pragma unroll
    for (int j = 0; j < 8; ++j) {
        int k = kb + 2 * j;
        b[2 * j]     = row[k];
        b[2 * j + 1] = row[k + 1];
    }
    return b;
}

// B fragment: 32x16 (KxN) from row-major matrix Bm[k][n], leading dim ld.
__device__ __forceinline__ v16bf load_B_frag(const __bf16* Bm, int ld, int lane) {
    int n  = lane & 15;
    int kb = (lane < 16) ? 0 : 16;
    v16bf b;
#pragma unroll
    for (int j = 0; j < 8; ++j) {
        int k = kb + 2 * j;
        b[2 * j]     = Bm[(size_t)k * ld + n];
        b[2 * j + 1] = Bm[(size_t)(k + 1) * ld + n];
    }
    return b;
}

// ---------------------------------------------------------------------------
// fp32 -> bf16 conversion (weights)
// ---------------------------------------------------------------------------
__global__ void f2bf_kernel(const float* __restrict__ src, __bf16* __restrict__ dst, int n) {
    int i = blockIdx.x * blockDim.x + threadIdx.x;
    if (i < n) dst[i] = f2bf(src[i]);
}

// ---------------------------------------------------------------------------
// Row LayerNorm: one block per row of D=512. Optional fp32 / bf16 outputs.
// ---------------------------------------------------------------------------
__global__ __launch_bounds__(256)
void ln_kernel(const float* __restrict__ x, const float* __restrict__ g,
               const float* __restrict__ bt, float* __restrict__ yf,
               __bf16* __restrict__ yb, int D) {
    int row = blockIdx.x;
    const float* xr = x + (size_t)row * D;
    __shared__ float red[32];
    float s = 0.f, s2 = 0.f;
    for (int i = threadIdx.x; i < D; i += blockDim.x) {
        float v = xr[i]; s += v; s2 += v * v;
    }
#pragma unroll
    for (int off = 16; off; off >>= 1) {
        s  += __shfl_xor(s, off);
        s2 += __shfl_xor(s2, off);
    }
    int wid = threadIdx.x >> 5;
    if ((threadIdx.x & 31) == 0) { red[wid] = s; red[8 + wid] = s2; }
    __syncthreads();
    if (threadIdx.x == 0) {
        float ts = 0.f, ts2 = 0.f;
        int nw = blockDim.x >> 5;
        for (int i = 0; i < nw; ++i) { ts += red[i]; ts2 += red[8 + i]; }
        float mean = ts / D;
        float var  = ts2 / D - mean * mean;
        red[16] = mean;
        red[17] = rsqrtf(var + 1e-5f);
    }
    __syncthreads();
    float mean = red[16], rstd = red[17];
    for (int i = threadIdx.x; i < D; i += blockDim.x) {
        float v = (xr[i] - mean) * rstd * g[i] + bt[i];
        if (yf) yf[(size_t)row * D + i] = v;
        if (yb) yb[(size_t)row * D + i] = f2bf(v);
    }
}

// ---------------------------------------------------------------------------
// bf16 WMMA GEMM: C[M,N] = act(A[M,K] @ W[N,K]^T + bias + residual)
// One wave per 32x64 output tile (2 A frags x 4 B frags = 8 WMMAs / k-step),
// 8 waves per block. act: 0=none, 1=exact GELU.
// Requires M % 32 == 0, N % 64 == 0, K % 32 == 0 (true for all call sites).
// ---------------------------------------------------------------------------
__global__ __launch_bounds__(256)
void gemm_bf16(const __bf16* __restrict__ A, const __bf16* __restrict__ W,
               const float* __restrict__ bias, const float* __restrict__ residual,
               float* __restrict__ Cf, __bf16* __restrict__ Cb,
               int M, int Nt, int K, int act) {
    int lane = threadIdx.x & 31;
    int wid  = threadIdx.x >> 5;
    int ntn   = Nt >> 6;                  // 64-wide column tiles
    int ntile = (M >> 5) * ntn;           // 32-tall row tiles
    int tile  = blockIdx.x * 8 + wid;
    if (tile >= ntile) return;            // wave-uniform (EXEC stays full)
    int tm = tile / ntn, tn = tile % ntn;
    const __bf16* Ab0 = A + (size_t)(tm * 32) * K;
    const __bf16* Ab1 = Ab0 + (size_t)16 * K;
    const __bf16* Wb  = W + (size_t)(tn * 64) * K;

    v8f acc[2][4] = {};
    for (int k0 = 0; k0 < K; k0 += 32) {
        if (k0 + 32 < K) {                // hint next k-slice (global_prefetch_b8)
            __builtin_prefetch(Ab0 + k0 + 32, 0, 3);
            __builtin_prefetch(Wb + k0 + 32, 0, 3);
        }
        v16bf a0 = load_A_frag(Ab0 + k0, K, lane);
        v16bf a1 = load_A_frag(Ab1 + k0, K, lane);
#pragma unroll
        for (int j = 0; j < 4; ++j) {
            v16bf b = load_Bt_frag(Wb + (size_t)(j * 16) * K + k0, K, lane);
            acc[0][j] = wmma_bf16(a0, b, acc[0][j]);
            acc[1][j] = wmma_bf16(a1, b, acc[1][j]);
        }
    }

    // C layout: lane -> col n = lane&15; rows mb+r (mb=0 lanes 0-15, mb=8 lanes 16-31)
    int n  = lane & 15;
    int mb = (lane < 16) ? 0 : 8;
#pragma unroll
    for (int g = 0; g < 2; ++g) {
#pragma unroll
        for (int j = 0; j < 4; ++j) {
            int col = tn * 64 + j * 16 + n;
            float bv = bias ? bias[col] : 0.0f;
#pragma unroll
            for (int r = 0; r < 8; ++r) {
                int row = tm * 32 + g * 16 + mb + r;
                float v = acc[g][j][r] + bv;
                if (residual) v += residual[(size_t)row * Nt + col];
                if (act) v = 0.5f * v * (1.0f + erff(v * 0.70710678118654752f));
                if (Cf) Cf[(size_t)row * Nt + col] = v;
                if (Cb) Cb[(size_t)row * Nt + col] = f2bf(v);
            }
        }
    }
}

// ---------------------------------------------------------------------------
// Flash attention (online softmax), 1 wave per 16-query tile per (b,h).
// DH=64. Handles length-masked cross-attn (lengths!=null, eps=1e-6 denom)
// and dense self-attn (lengths==null, fixed_len).
// ---------------------------------------------------------------------------
__global__ __launch_bounds__(32)
void attn_kernel(const __bf16* __restrict__ Qm, long long qbs, int qrs,
                 const __bf16* __restrict__ Km, long long kbs, int krs,
                 const __bf16* __restrict__ Vm, long long vbs, int vrs,
                 __bf16* __restrict__ Ob, long long obs, int ors,
                 const int* __restrict__ lengths, int fixed_len,
                 float scale, float eps) {
    int lane = threadIdx.x;
    int qt = blockIdx.x, h = blockIdx.y, b = blockIdx.z;
    int len = lengths ? lengths[b] : fixed_len;

    const __bf16* q = Qm + (size_t)b * qbs + (size_t)(qt * 16) * qrs + h * DHc;
    v16bf qa0 = load_A_frag(q, qrs, lane);
    v16bf qa1 = load_A_frag(q + 32, qrs, lane);

    const float NEG_INF = -__builtin_inff();
    float mrun[8], lrun[8];
#pragma unroll
    for (int r = 0; r < 8; ++r) { mrun[r] = NEG_INF; lrun[r] = 0.f; }
    v8f o0 = {}, o1 = {}, o2 = {}, o3 = {};

    __shared__ __bf16 Pt[16 * 32];
    int n  = lane & 15;
    int mb = (lane < 16) ? 0 : 8;

    for (int n0 = 0; n0 < len; n0 += 32) {
        const __bf16* kp = Km + (size_t)b * kbs + (size_t)n0 * krs + h * DHc;
        // scores S[16q x 32k]: two 16x16 tiles, each = Q(16x64) @ K^T via 2 WMMAs
        v16bf b0 = load_Bt_frag(kp, krs, lane);
        v16bf b1 = load_Bt_frag(kp + 32, krs, lane);
        v16bf b2 = load_Bt_frag(kp + (size_t)16 * krs, krs, lane);
        v16bf b3 = load_Bt_frag(kp + (size_t)16 * krs + 32, krs, lane);
        v8f s0 = {}, s1 = {};
        s0 = wmma_bf16(qa0, b0, s0); s0 = wmma_bf16(qa1, b1, s0);
        s1 = wmma_bf16(qa0, b2, s1); s1 = wmma_bf16(qa1, b3, s1);

        bool ok0 = (n0 + n) < len;
        bool ok1 = (n0 + 16 + n) < len;
        float alpha[8];
#pragma unroll
        for (int r = 0; r < 8; ++r) {
            float a0 = ok0 ? s0[r] * scale : NEG_INF;
            float a1 = ok1 ? s1[r] * scale : NEG_INF;
            // row max across the 16 lanes holding this row's columns
            float mx = fmaxf(a0, a1);
            mx = fmaxf(mx, __shfl_xor(mx, 1));
            mx = fmaxf(mx, __shfl_xor(mx, 2));
            mx = fmaxf(mx, __shfl_xor(mx, 4));
            mx = fmaxf(mx, __shfl_xor(mx, 8));
            float mn = fmaxf(mrun[r], mx);
            float msafe = (mn == NEG_INF) ? 0.f : mn;            // reference safe_m
            alpha[r] = (mrun[r] == NEG_INF) ? 0.f : __expf(mrun[r] - mn);
            float p0 = __expf(a0 - msafe);                        // -inf -> 0
            float p1 = __expf(a1 - msafe);
            float rs = p0 + p1;
            rs += __shfl_xor(rs, 1);
            rs += __shfl_xor(rs, 2);
            rs += __shfl_xor(rs, 4);
            rs += __shfl_xor(rs, 8);
            lrun[r] = lrun[r] * alpha[r] + rs;
            mrun[r] = mn;
            o0[r] *= alpha[r]; o1[r] *= alpha[r];
            o2[r] *= alpha[r]; o3[r] *= alpha[r];
            Pt[(mb + r) * 32 + n]      = f2bf(p0);
            Pt[(mb + r) * 32 + 16 + n] = f2bf(p1);
        }
        __syncthreads();
        // O += P(16x32) @ V(32x64): A from LDS, 4 B fragments from V rows
        v16bf pa = load_A_frag(Pt, 32, lane);
        const __bf16* vp = Vm + (size_t)b * vbs + (size_t)n0 * vrs + h * DHc;
        v16bf vb0 = load_B_frag(vp, vrs, lane);
        v16bf vb1 = load_B_frag(vp + 16, vrs, lane);
        v16bf vb2 = load_B_frag(vp + 32, vrs, lane);
        v16bf vb3 = load_B_frag(vp + 48, vrs, lane);
        o0 = wmma_bf16(pa, vb0, o0);
        o1 = wmma_bf16(pa, vb1, o1);
        o2 = wmma_bf16(pa, vb2, o2);
        o3 = wmma_bf16(pa, vb3, o3);
        __syncthreads();
    }
#pragma unroll
    for (int r = 0; r < 8; ++r) {
        float inv = 1.0f / (lrun[r] + eps);
        size_t base = (size_t)b * obs + (size_t)(qt * 16 + mb + r) * ors + h * DHc + n;
        Ob[base]      = f2bf(o0[r] * inv);
        Ob[base + 16] = f2bf(o1[r] * inv);
        Ob[base + 32] = f2bf(o2[r] * inv);
        Ob[base + 48] = f2bf(o3[r] * inv);
    }
}

// ---------------------------------------------------------------------------
// Broadcast queries [L,D] -> x [B,L,D]
// ---------------------------------------------------------------------------
__global__ void bcast_kernel(const float* __restrict__ q, float* __restrict__ x,
                             int LD, int total) {
    int i = blockIdx.x * blockDim.x + threadIdx.x;
    if (i < total) x[i] = q[i % LD];
}

// ---------------------------------------------------------------------------
// out[b,d] = mean over l of xf[b,l,d]
// ---------------------------------------------------------------------------
__global__ void mean_kernel(const float* __restrict__ xf, float* __restrict__ out,
                            int Lr, int D, int total) {
    int idx = blockIdx.x * blockDim.x + threadIdx.x;
    if (idx >= total) return;
    int b = idx / D, d = idx % D;
    float s = 0.f;
    for (int l = 0; l < Lr; ++l) s += xf[((size_t)b * Lr + l) * D + d];
    out[idx] = s / (float)Lr;
}

// ---------------------------------------------------------------------------
// Host orchestration
// ---------------------------------------------------------------------------
extern "C" void kernel_launch(void* const* d_in, const int* in_sizes, int n_in,
                              void* d_out, int out_size, void* d_ws, size_t ws_size,
                              hipStream_t stream) {
    const float* tokens   = (const float*)d_in[0];
    const float* latents  = (const float*)d_in[1];
    const float* lnq_g    = (const float*)d_in[2];
    const float* lnq_b    = (const float*)d_in[3];
    const float* lnkv_g   = (const float*)d_in[4];
    const float* lnkv_b   = (const float*)d_in[5];
    const float* Wq       = (const float*)d_in[6];
    const float* Wkv      = (const float*)d_in[7];
    const float* Wo       = (const float*)d_in[8];
    const float* bo       = (const float*)d_in[9];
    const float* bln1g    = (const float*)d_in[10];
    const float* bln1b    = (const float*)d_in[11];
    const float* binw     = (const float*)d_in[12];
    const float* binb     = (const float*)d_in[13];
    const float* bow      = (const float*)d_in[14];
    const float* bob      = (const float*)d_in[15];
    const float* bln2g    = (const float*)d_in[16];
    const float* bln2b    = (const float*)d_in[17];
    const float* bw1      = (const float*)d_in[18];
    const float* bb1      = (const float*)d_in[19];
    const float* bw2      = (const float*)d_in[20];
    const float* bb2      = (const float*)d_in[21];
    const float* fin_g    = (const float*)d_in[22];
    const float* fin_b    = (const float*)d_in[23];
    const int*   lengths  = (const int*)d_in[24];

    char* ws = (char*)d_ws;
    size_t off = 0;
    auto take = [&](size_t bytes) -> char* {
        char* p = ws + off;
        off += (bytes + 255) & ~(size_t)255;
        return p;
    };

    float*  qf       = (float*)take((size_t)Lc * Dc * 4);
    __bf16* qbf      = (__bf16*)take((size_t)Lc * Dc * 2);
    __bf16* qproj_bf = (__bf16*)take((size_t)Lc * Dc * 2);
    __bf16* tok_bf   = (__bf16*)take((size_t)Bc * Nc * Dc * 2);
    __bf16* kv_bf    = (__bf16*)take((size_t)Bc * Nc * 2 * Dc * 2);
    __bf16* attn_bf  = (__bf16*)take((size_t)Bc * Lc * Dc * 2);
    float*  x        = (float*)take((size_t)Bc * Lc * Dc * 4);
    __bf16* hbf      = (__bf16*)take((size_t)Bc * Lc * Dc * 2);
    __bf16* qkvbf    = (__bf16*)take((size_t)Bc * Lc * 3 * Dc * 2);
    __bf16* ffbf     = (__bf16*)take((size_t)Bc * Lc * 4 * Dc * 2);
    float*  xf       = (float*)take((size_t)Bc * Lc * Dc * 4);
    __bf16* wq_bf    = (__bf16*)take((size_t)Dc * Dc * 2);
    __bf16* wkv_bf   = (__bf16*)take((size_t)2 * Dc * Dc * 2);
    __bf16* wo_bf    = (__bf16*)take((size_t)Dc * Dc * 2);
    __bf16* inw_bf   = (__bf16*)take((size_t)BKS * 3 * Dc * Dc * 2);
    __bf16* ow_bf    = (__bf16*)take((size_t)BKS * Dc * Dc * 2);
    __bf16* w1_bf    = (__bf16*)take((size_t)BKS * 4 * Dc * Dc * 2);
    __bf16* w2_bf    = (__bf16*)take((size_t)BKS * Dc * 4 * Dc * 2);

    auto cvt = [&](const float* s, __bf16* d, int n) {
        f2bf_kernel<<<(n + 255) / 256, 256, 0, stream>>>(s, d, n);
    };
    cvt(Wq, wq_bf, Dc * Dc);
    cvt(Wkv, wkv_bf, 2 * Dc * Dc);
    cvt(Wo, wo_bf, Dc * Dc);
    cvt(binw, inw_bf, BKS * 3 * Dc * Dc);
    cvt(bow, ow_bf, BKS * Dc * Dc);
    cvt(bw1, w1_bf, BKS * 4 * Dc * Dc);
    cvt(bw2, w2_bf, BKS * Dc * 4 * Dc);

    // queries = LN(latents) (identical across batch); tokens_norm = LN(tokens)
    ln_kernel<<<Lc, 256, 0, stream>>>(latents, lnq_g, lnq_b, qf, qbf, Dc);
    ln_kernel<<<Bc * Nc, 256, 0, stream>>>(tokens, lnkv_g, lnkv_b, nullptr, tok_bf, Dc);

    auto gemm = [&](const __bf16* A, const __bf16* W, const float* bias,
                    const float* res, float* Cf, __bf16* Cb,
                    int M, int Nt, int K, int act) {
        int tiles = (M / 32) * (Nt / 64);
        gemm_bf16<<<(tiles + 7) / 8, 256, 0, stream>>>(A, W, bias, res, Cf, Cb, M, Nt, K, act);
    };

    // Q = queries @ Wq^T ;  KV = tokens_norm @ Wkv^T  (dominant GEMM)
    gemm(qbf, wq_bf, nullptr, nullptr, nullptr, qproj_bf, Lc, Dc, Dc, 0);
    gemm(tok_bf, wkv_bf, nullptr, nullptr, nullptr, kv_bf, Bc * Nc, 2 * Dc, Dc, 0);

    // x = broadcast(queries)
    {
        int total = Bc * Lc * Dc;
        bcast_kernel<<<(total + 255) / 256, 256, 0, stream>>>(qf, x, Lc * Dc, total);
    }

    // masked cross attention (flash): K stride 2D within kv_bf, V at +D
    attn_kernel<<<dim3(Lc / 16, Hc, Bc), 32, 0, stream>>>(
        qproj_bf, 0LL, Dc,
        kv_bf, (long long)Nc * 2 * Dc, 2 * Dc,
        kv_bf + Dc, (long long)Nc * 2 * Dc, 2 * Dc,
        attn_bf, (long long)Lc * Dc, Dc,
        lengths, 0, 0.125f, 1e-6f);

    // x = queries + attn @ Wo^T + bo
    gemm(attn_bf, wo_bf, bo, x, x, nullptr, Bc * Lc, Dc, Dc, 0);

    for (int i = 0; i < BKS; ++i) {
        // h1 = LN(x)
        ln_kernel<<<Bc * Lc, 256, 0, stream>>>(x, bln1g + i * Dc, bln1b + i * Dc,
                                               nullptr, hbf, Dc);
        // qkv = h1 @ inw^T + inb
        gemm(hbf, inw_bf + (size_t)i * 3 * Dc * Dc, binb + i * 3 * Dc,
             nullptr, nullptr, qkvbf, Bc * Lc, 3 * Dc, Dc, 0);
        // dense self-attention over L=256
        attn_kernel<<<dim3(Lc / 16, Hc, Bc), 32, 0, stream>>>(
            qkvbf, (long long)Lc * 3 * Dc, 3 * Dc,
            qkvbf + Dc, (long long)Lc * 3 * Dc, 3 * Dc,
            qkvbf + 2 * Dc, (long long)Lc * 3 * Dc, 3 * Dc,
            attn_bf, (long long)Lc * Dc, Dc,
            nullptr, Lc, 0.125f, 0.0f);
        // x += attn @ ow^T + ob
        gemm(attn_bf, ow_bf + (size_t)i * Dc * Dc, bob + i * Dc,
             x, x, nullptr, Bc * Lc, Dc, Dc, 0);
        // h2 = LN(x)
        ln_kernel<<<Bc * Lc, 256, 0, stream>>>(x, bln2g + i * Dc, bln2b + i * Dc,
                                               nullptr, hbf, Dc);
        // f = gelu(h2 @ w1^T + b1)
        gemm(hbf, w1_bf + (size_t)i * 4 * Dc * Dc, bb1 + i * 4 * Dc,
             nullptr, nullptr, ffbf, Bc * Lc, 4 * Dc, Dc, 1);
        // x += f @ w2^T + b2
        gemm(ffbf, w2_bf + (size_t)i * Dc * 4 * Dc, bb2 + i * Dc,
             x, x, nullptr, Bc * Lc, Dc, 4 * Dc, 0);
    }

    // final LN, then mean over L
    ln_kernel<<<Bc * Lc, 256, 0, stream>>>(x, fin_g, fin_b, xf, nullptr, Dc);
    {
        int total = Bc * Dc;
        mean_kernel<<<(total + 255) / 256, 256, 0, stream>>>(xf, (float*)d_out, Lc, Dc, total);
    }
}
